// FuNet_19533511262603
// MI455X (gfx1250) — compile-verified, matrix operand order checked by reading, not run on hardware
//
#include <hip/hip_runtime.h>
#include <hip/hip_bf16.h>

typedef __attribute__((ext_vector_type(16))) _Float16 v16h;
typedef __attribute__((ext_vector_type(8)))  float    v8f;

union FragU32 { v16h h; unsigned int u[8]; };
union FragH16 { v16h h; _Float16 e[16]; };

// -------------------------------------------------------------------------
// f32 -> f16 convert/copy (also used to build channel-concat buffers)
// -------------------------------------------------------------------------
__global__ __launch_bounds__(256) void cvt_f32_f16(const float* __restrict__ src,
                                                   _Float16* __restrict__ dst,
                                                   size_t n) {
    size_t i = (size_t)blockIdx.x * 256 + threadIdx.x;
    if (i < n) dst[i] = (_Float16)src[i];
}

// -------------------------------------------------------------------------
// Implicit-GEMM conv via WMMA + async global->LDS staging.
//   GEMM: M = Cout (MT tiles of 16 per wave), N = pixels (128/block, 16/wave),
//   K = Cin*KS*KS, ordering k = tap*CIN + ci  (tap = kh*KS + kw)
//   LDS: A tile  [MT*16][K]         (f16, contiguous K)
//        patch   [kh][ci][CWP]      (f16, column direction contiguous ->
//                                    DMA'd with global_load_async_to_lds_b128)
//   Each wave reuses one B fragment for MT WMMAs (B-gather amortization).
// ACT: 0 = none, 1 = LeakyReLU(0.1), 2 = GELU(erf)
// -------------------------------------------------------------------------
template <int CIN, int KS, int ACT, bool OUT_F32, bool HAS_BIAS, int MT = 2>
__global__ __launch_bounds__(256) void conv_wmma(
    const _Float16* __restrict__ in,    // [B][CIN][H][W]
    const _Float16* __restrict__ wgt,   // [Cout][CIN][KS][KS] (f16)
    const float*    __restrict__ bias,  // [Cout] or nullptr
    void*           __restrict__ outp,  // f16 or f32: [B][coutTot][H][W]
    int coutTot, int H, int W)
{
    constexpr int NT   = 128;
    constexpr int PAD  = KS / 2;
    constexpr int POFF = (KS == 3) ? 7 : 0;     // patch col c stored at c+POFF
    constexpr int CWP  = (KS == 3) ? 144 : 128; // LDS row stride (halves)
    constexpr int K    = CIN * KS * KS;

    extern __shared__ float4 smem4[];
    _Float16* aS = (_Float16*)smem4;             // MT*16*K halves
    _Float16* pS = aS + MT * 16 * K;             // KS*CIN*CWP halves

    const int HW = H * W;
    const int tilesPerRow = W / NT;
    const int bx = blockIdx.x;
    const int h  = bx / tilesPerRow;
    const int w0 = (bx - h * tilesPerRow) * NT;
    const int coBase = blockIdx.y * (16 * MT);
    const int b   = blockIdx.z;
    const int tid = threadIdx.x;

    // Prefetch weight tile stream (gfx1250 global_prefetch_b8)
    __builtin_prefetch(wgt + (size_t)coBase * CIN * KS * KS, 0, 0);

    // ---- stage A (weights) into LDS with K = tap*CIN + ci ordering ----
    for (int i = tid; i < MT * 16 * K; i += 256) {
        int m   = i / K;
        int k   = i - m * K;
        int tap = k / CIN;
        int ci  = k - tap * CIN;
        aS[i] = wgt[((size_t)(coBase + m) * CIN + ci) * (KS * KS) + tap];
    }

    // ---- stage input patch via async DMA: [kh][ci][CWP] ----
    const _Float16* inb = in + (size_t)b * CIN * HW;
    const unsigned int pBase = (unsigned int)(size_t)pS;  // LDS byte offset
    {
        // central 128 cols of each (kh, ci) row: 16 x b128 async transfers
        const int T = KS * CIN * 16;
        for (int i = tid; i < T; i += 256) {
            int kh = i / (CIN * 16);
            int r  = i - kh * (CIN * 16);
            int ci = r >> 4;
            int j  = r & 15;
            int hs = h + kh - PAD;
            // patch col = PAD + 8*j .. +7  ->  pos = PAD + POFF + 8*j (16B aligned)
            unsigned int ldsb =
                pBase + (unsigned int)((((kh * CIN + ci) * CWP) + PAD + POFF + 8 * j) * 2);
            if ((unsigned)hs < (unsigned)H) {
                unsigned long long ga = (unsigned long long)
                    (inb + (size_t)ci * HW + (size_t)hs * W + w0 + 8 * j);
                asm volatile("global_load_async_to_lds_b128 %0, %1, off"
                             :: "v"(ldsb), "v"(ga) : "memory");
            } else {
                float4 z; z.x = z.y = z.z = z.w = 0.0f;
                *(float4*)(pS + ((kh * CIN + ci) * CWP) + PAD + POFF + 8 * j) = z;
            }
        }
        if constexpr (KS == 3) {
            // halo columns (patch col 0 and NT+1), zero-padded at image edges
            for (int e = tid; e < 3 * CIN * 2; e += 256) {
                int kh   = e / (CIN * 2);
                int r    = e - kh * (CIN * 2);
                int ci   = r >> 1;
                int side = r & 1;
                int col  = side ? (NT + 1) : 0;
                int hs   = h + kh - 1;
                int wsr  = w0 + col - 1;
                _Float16 v = (_Float16)0.0f;
                if ((unsigned)hs < (unsigned)H && (unsigned)wsr < (unsigned)W)
                    v = inb[(size_t)ci * HW + (size_t)hs * W + wsr];
                pS[(kh * CIN + ci) * CWP + col + POFF] = v;
            }
        }
    }
    asm volatile("s_wait_asynccnt 0x0" ::: "memory");
    __syncthreads();

    const int wave  = tid >> 5;
    const int lane  = tid & 31;
    const int nOff  = wave * 16;
    const int laneN = lane & 15;
    const int hi    = lane >> 4;      // 0: K {0-7,16-23}  1: K {8-15,24-31}

    v8f acc[MT];
#pragma unroll
    for (int t = 0; t < MT; ++t) acc[t] = {};

    const unsigned int* a32 = (const unsigned int*)aS;

    for (int k0 = 0; k0 < K; k0 += 32) {
        // B fragment (shared across MT tiles): tap constant within 32-K block
        int tap = k0 / CIN;
        int ci0 = k0 - tap * CIN;
        int kh  = tap / KS;
        int kw  = tap - kh * KS;
        const _Float16* bp =
            pS + (kh * CIN + ci0 + hi * 8) * CWP + (nOff + laneN + kw + POFF);
        FragH16 bf;
#pragma unroll
        for (int j = 0; j < 8; ++j) {
            bf.e[j]     = bp[(size_t)j * CWP];
            bf.e[8 + j] = bp[(size_t)(16 + j) * CWP];
        }
        // MT A fragments, one WMMA each, reusing bf
        FragU32 a[MT];
#pragma unroll
        for (int t = 0; t < MT; ++t) {
            int abase = ((t * 16 + laneN) * K + k0 + hi * 8) >> 1;
#pragma unroll
            for (int j = 0; j < 4; ++j) {
                a[t].u[j]     = a32[abase + j];
                a[t].u[4 + j] = a32[abase + 8 + j];
            }
        }
#pragma unroll
        for (int t = 0; t < MT; ++t)
            acc[t] = __builtin_amdgcn_wmma_f32_16x16x32_f16(
                false, a[t].h, false, bf.h, (short)0, acc[t], false, false);
    }

    // ---- epilogue: bias + activation + store (C layout: n=lane&15, m=r+8*hi)
    const int px = h * W + w0 + nOff + laneN;
#pragma unroll
    for (int t = 0; t < MT; ++t) {
#pragma unroll
        for (int r = 0; r < 8; ++r) {
            int   m  = t * 16 + r + hi * 8;
            int   co = coBase + m;
            float v  = acc[t][r];
            if constexpr (HAS_BIAS) v += bias[co];
            if constexpr (ACT == 1) v = (v >= 0.0f) ? v : 0.1f * v;
            if constexpr (ACT == 2) v = 0.5f * v * (1.0f + erff(v * 0.70710678118654752f));
            size_t oi = ((size_t)b * coutTot + co) * HW + px;
            if constexpr (OUT_F32) ((float*)outp)[oi] = v;
            else                   ((_Float16*)outp)[oi] = (_Float16)v;
        }
    }
}

// -------------------------------------------------------------------------
// Per-pixel: corr over 9 candidates with x, top-3 (desc, ties->low idx),
// gather selected candidates -> pre[B][32][3][H][W]
// -------------------------------------------------------------------------
__global__ __launch_bounds__(256) void corr_topk_gather(
    const float* __restrict__ x,       // [B][32][HW] f32
    const _Float16* __restrict__ warp, // [B][288][HW] f16 (co = c*9 + o)
    _Float16* __restrict__ pre,        // [B][32][3][HW] f16
    int HW)
{
    int p = blockIdx.x * 256 + threadIdx.x;
    if (p >= 2 * HW) return;
    int b = p / HW;
    int q = p - b * HW;

    const float*    xb = x    + (size_t)b * 32  * HW + q;
    const _Float16* wb = warp + (size_t)b * 288 * HW + q;

    float corr[9];
#pragma unroll
    for (int o = 0; o < 9; ++o) corr[o] = 0.0f;

    for (int c = 0; c < 32; ++c) {
        float xv = xb[(size_t)c * HW];
#pragma unroll
        for (int o = 0; o < 9; ++o)
            corr[o] += (float)wb[(size_t)(c * 9 + o) * HW] * xv;
    }

    int idx[3];
    bool used[9];
#pragma unroll
    for (int o = 0; o < 9; ++o) used[o] = false;
    for (int k = 0; k < 3; ++k) {
        int bi = 0; float bv = -3.0e38f;
#pragma unroll
        for (int o = 0; o < 9; ++o)
            if (!used[o] && corr[o] > bv) { bv = corr[o]; bi = o; }
        used[bi] = true;
        idx[k] = bi;
    }

    _Float16* pb = pre + (size_t)b * 96 * HW + q;
    for (int c = 0; c < 32; ++c) {
#pragma unroll
        for (int k = 0; k < 3; ++k)
            pb[(size_t)(c * 3 + k) * HW] = wb[(size_t)(c * 9 + idx[k]) * HW];
    }
}

// -------------------------------------------------------------------------
// Selection conv: (1,3,3,3) over the 3 topk slots, per (b,c) plane.
// -------------------------------------------------------------------------
__global__ __launch_bounds__(256) void sel_conv(
    const _Float16* __restrict__ pre,   // [B*32][3][H][W]
    const float* __restrict__ wsel,     // [27]
    const float* __restrict__ bsel,     // [1]
    _Float16* __restrict__ out,         // [B*32][H][W]
    int H, int W)
{
    __shared__ float sw[28];
    if (threadIdx.x < 27) sw[threadIdx.x] = wsel[threadIdx.x];
    if (threadIdx.x == 27) sw[27] = bsel[0];
    __syncthreads();

    int HW  = H * W;
    int gid = blockIdx.x * 256 + threadIdx.x;
    if (gid >= 2 * 32 * HW) return;
    int bc = gid / HW;
    int q  = gid - bc * HW;
    int h  = q / W;
    int w  = q - h * W;

    const _Float16* pb = pre + (size_t)bc * 3 * HW;
    float acc = sw[27];
#pragma unroll
    for (int k = 0; k < 3; ++k)
#pragma unroll
        for (int kh = 0; kh < 3; ++kh) {
            int hs = h + kh - 1;
            if ((unsigned)hs >= (unsigned)H) continue;
#pragma unroll
            for (int kw = 0; kw < 3; ++kw) {
                int wsr = w + kw - 1;
                if ((unsigned)wsr >= (unsigned)W) continue;
                acc += sw[(k * 3 + kh) * 3 + kw] *
                       (float)pb[(size_t)k * HW + (size_t)hs * W + wsr];
            }
        }
    out[gid] = (_Float16)acc;
}

// -------------------------------------------------------------------------
extern "C" void kernel_launch(void* const* d_in, const int* in_sizes, int n_in,
                              void* d_out, int out_size, void* d_ws, size_t ws_size,
                              hipStream_t stream) {
    const float* x      = (const float*)d_in[0];
    const float* key    = (const float*)d_in[1];
    const float* w_off1 = (const float*)d_in[2];
    const float* b_off1 = (const float*)d_in[3];
    const float* w_off2 = (const float*)d_in[4];
    const float* b_off2 = (const float*)d_in[5];
    const float* w_dcn  = (const float*)d_in[6];
    const float* b_dcn  = (const float*)d_in[7];
    const float* w_sel  = (const float*)d_in[8];
    const float* b_sel  = (const float*)d_in[9];
    const float* w_t1   = (const float*)d_in[10];
    const float* w_t2   = (const float*)d_in[11];
    const float* w_t3   = (const float*)d_in[12];
    const float* w_t4   = (const float*)d_in[13];
    const float* w_t5   = (const float*)d_in[14];
    float* out = (float*)d_out;

    const int H = 256, W = 256;
    const size_t HW = (size_t)H * W;

    // ---- workspace layout (f16 elements) ----
    _Float16* ws = (_Float16*)d_ws;
    size_t off = 0;
    auto alloc = [&](size_t n) { _Float16* p = ws + off; off += n; return p; };
    _Float16* cat1h = alloc(2 * 64 * HW);    // [x | key]
    _Float16* cat2h = alloc(2 * 64 * HW);    // [key | off2]
    _Float16* offh  = alloc(2 * 32 * HW);
    _Float16* warph = alloc(2 * 288 * HW);
    _Float16* preh  = alloc(2 * 96 * HW);
    _Float16* wo1h  = alloc(32 * 64 * 9);
    _Float16* wo2h  = alloc(32 * 32 * 9);
    _Float16* wdch  = alloc(288 * 64 * 9);
    _Float16* wt1h  = alloc(32 * 32 * 9);
    _Float16* wt2h  = alloc(128 * 32);
    _Float16* wt3h  = alloc(128 * 128 * 9);
    _Float16* wt4h  = alloc(32 * 128);
    _Float16* wt5h  = alloc(32 * 32 * 9);
    // lifetime-based aliases for tail activations
    _Float16* keywh = cat2h;                  // dead after dcn conv
    _Float16* y1h   = offh;                   // dead after conv2
    _Float16* y2h   = warph;                  // dead after gather
    _Float16* y3h   = warph + 2 * 128 * HW;   // fits in warph region
    _Float16* y4h   = cat1h;                  // dead after conv1

    auto cvt = [&](const float* s, _Float16* d, size_t n) {
        cvt_f32_f16<<<(unsigned)((n + 255) / 256), 256, 0, stream>>>(s, d, n);
    };

    // ---- build f16 inputs / weights ----
    cvt(x,             cat1h,               32 * HW);   // b0 x
    cvt(x + 32 * HW,   cat1h + 64 * HW,     32 * HW);   // b1 x
    cvt(key,           cat1h + 32 * HW,     32 * HW);   // b0 key
    cvt(key + 32 * HW, cat1h + 96 * HW,     32 * HW);   // b1 key
    cvt(key,           cat2h,               32 * HW);   // b0 key
    cvt(key + 32 * HW, cat2h + 64 * HW,     32 * HW);   // b1 key
    cvt(w_off1, wo1h, 32 * 64 * 9);
    cvt(w_off2, wo2h, 32 * 32 * 9);
    cvt(w_dcn,  wdch, 288 * 64 * 9);
    cvt(w_t1,   wt1h, 32 * 32 * 9);
    cvt(w_t2,   wt2h, 128 * 32);
    cvt(w_t3,   wt3h, 128 * 128 * 9);
    cvt(w_t4,   wt4h, 32 * 128);
    cvt(w_t5,   wt5h, 32 * 32 * 9);

    // ---- conv configs ----
    const dim3 blk(256);
    const int tiles = (W / 128) * H;          // 512 pixel tiles per image
    auto shb = [](int cin, int ks) {
        int K   = cin * ks * ks;
        int cwp = (ks == 3) ? 144 : 128;
        return (size_t)(2 * 16 * K + ks * cin * cwp) * sizeof(_Float16);
    };
    const size_t sh_64_3  = shb(64, 3);
    const size_t sh_32_3  = shb(32, 3);
    const size_t sh_128_3 = shb(128, 3);
    const size_t sh_32_1  = shb(32, 1);
    const size_t sh_128_1 = shb(128, 1);

    hipFuncSetAttribute((const void*)conv_wmma<64, 3, 1, false, true>,
                        hipFuncAttributeMaxDynamicSharedMemorySize, (int)sh_64_3);
    hipFuncSetAttribute((const void*)conv_wmma<32, 3, 1, false, true>,
                        hipFuncAttributeMaxDynamicSharedMemorySize, (int)sh_32_3);
    hipFuncSetAttribute((const void*)conv_wmma<64, 3, 0, false, true>,
                        hipFuncAttributeMaxDynamicSharedMemorySize, (int)sh_64_3);
    hipFuncSetAttribute((const void*)conv_wmma<32, 3, 2, false, false>,
                        hipFuncAttributeMaxDynamicSharedMemorySize, (int)sh_32_3);
    hipFuncSetAttribute((const void*)conv_wmma<32, 1, 2, false, false>,
                        hipFuncAttributeMaxDynamicSharedMemorySize, (int)sh_32_1);
    hipFuncSetAttribute((const void*)conv_wmma<128, 3, 2, false, false>,
                        hipFuncAttributeMaxDynamicSharedMemorySize, (int)sh_128_3);
    hipFuncSetAttribute((const void*)conv_wmma<128, 1, 2, false, false>,
                        hipFuncAttributeMaxDynamicSharedMemorySize, (int)sh_128_1);
    hipFuncSetAttribute((const void*)conv_wmma<32, 3, 0, true, false>,
                        hipFuncAttributeMaxDynamicSharedMemorySize, (int)sh_32_3);

    // offset branch: lrelu(conv(cat[x,key], w_off1)+b)   [Cout=32 -> y=1]
    conv_wmma<64, 3, 1, false, true><<<dim3(tiles, 1, 2), blk, sh_64_3, stream>>>(
        cat1h, wo1h, b_off1, offh, 32, H, W);
    // off2 -> second half of cat2 (coutTot = 64, base offset 32*HW)
    conv_wmma<32, 3, 1, false, true><<<dim3(tiles, 1, 2), blk, sh_32_3, stream>>>(
        offh, wo2h, b_off2, cat2h + 32 * HW, 64, H, W);
    // dcn proxy conv: cat[key,off] -> 288 candidate channels  [288/32 = 9]
    conv_wmma<64, 3, 0, false, true><<<dim3(tiles, 9, 2), blk, sh_64_3, stream>>>(
        cat2h, wdch, b_dcn, warph, 288, H, W);

    // corr + top3 + gather
    corr_topk_gather<<<(unsigned)((2 * HW + 255) / 256), blk, 0, stream>>>(
        x, warph, preh, (int)HW);
    // selection conv over topk slots
    sel_conv<<<(unsigned)((2 * 32 * HW + 255) / 256), blk, 0, stream>>>(
        preh, w_sel, b_sel, keywh, H, W);

    // tail: conv-GELU bottleneck stack (activation fused at producer store)
    conv_wmma<32, 3, 2, false, false><<<dim3(tiles, 1, 2), blk, sh_32_3, stream>>>(
        keywh, wt1h, nullptr, y1h, 32, H, W);
    conv_wmma<32, 1, 2, false, false><<<dim3(tiles, 4, 2), blk, sh_32_1, stream>>>(
        y1h, wt2h, nullptr, y2h, 128, H, W);
    conv_wmma<128, 3, 2, false, false><<<dim3(tiles, 4, 2), blk, sh_128_3, stream>>>(
        y2h, wt3h, nullptr, y3h, 128, H, W);
    conv_wmma<128, 1, 2, false, false><<<dim3(tiles, 1, 2), blk, sh_128_1, stream>>>(
        y3h, wt4h, nullptr, y4h, 32, H, W);
    conv_wmma<32, 3, 0, true, false><<<dim3(tiles, 1, 2), blk, sh_32_3, stream>>>(
        y4h, wt5h, nullptr, out, 32, H, W);

    (void)in_sizes; (void)n_in; (void)out_size; (void)ws_size;
}
